// MoEAttention_15204184227987
// MI455X (gfx1250) — matrix operand order repaired and checked by reading, not compiled, not run on hardware
//
#include <hip/hip_runtime.h>

#define D_MODEL 1024
#define N_HEADS 16
#define HEAD_DIM 64
#define NUM_EXPERTS 8
#define BB 4
#define TT 2048

typedef __attribute__((ext_vector_type(16))) __bf16 v16bf;
typedef __attribute__((ext_vector_type(8)))  float  v8f;
typedef __attribute__((ext_vector_type(4)))  int    v4i;

union FragB16 { v16bf v; unsigned u[8]; };

#define AS1 __attribute__((address_space(1)))
#define AS3 __attribute__((address_space(3)))

#if defined(__has_builtin)
#if __has_builtin(__builtin_amdgcn_global_load_async_to_lds_b128)
#define HAVE_ASYNC_LDS 1
#endif
#endif
#ifndef HAVE_ASYNC_LDS
#define HAVE_ASYNC_LDS 0
#endif

#if HAVE_ASYNC_LDS
// Async DMA 16B global -> LDS. Imm offset applies to BOTH addresses (ISA 8/4.4).
// Signature (from clang diagnostic): (int4 AS1*, int4 AS3*, imm int, imm int)
#define ASYNC_CP16(gp, lp, off)                                              \
  __builtin_amdgcn_global_load_async_to_lds_b128((AS1 v4i*)(gp),             \
                                                 (AS3 v4i*)(lp), (off), 0)
__device__ __forceinline__ void wait_async0() {
#if __has_builtin(__builtin_amdgcn_s_wait_asynccnt)
  __builtin_amdgcn_s_wait_asynccnt(0);
#else
  asm volatile("s_wait_asynccnt 0x0" ::: "memory");
#endif
}
#endif

__device__ __forceinline__ unsigned short f2bf(float f) {
  unsigned u = __float_as_uint(f);
  unsigned r = 0x7FFFu + ((u >> 16) & 1u);
  return (unsigned short)((u + r) >> 16);
}
__device__ __forceinline__ unsigned pack2bf(float a, float b) {
  return (unsigned)f2bf(a) | ((unsigned)f2bf(b) << 16);
}

// ---------------------------------------------------------------------------
// Kernel 1: gating (mean pool -> relu MLP -> softmax) + effective biases
// ---------------------------------------------------------------------------
__global__ __launch_bounds__(256) void gate_kernel(
    const float* __restrict__ x,
    const float* __restrict__ Wg1, const float* __restrict__ bg1,
    const float* __restrict__ Wg2, const float* __restrict__ bg2,
    const float* __restrict__ bq, const float* __restrict__ bk,
    const float* __restrict__ bv,
    float* __restrict__ gw, float* __restrict__ beff)
{
  __shared__ float pooled[D_MODEL];
  __shared__ float gh[128];
  __shared__ float gwl[NUM_EXPERTS];
  const int b = blockIdx.x, tid = threadIdx.x;

  for (int c = tid; c < D_MODEL; c += 256) {
    float s = 0.0f;
    const float* xp = x + (size_t)b * TT * D_MODEL + c;
    for (int t = 0; t < TT; ++t) s += xp[(size_t)t * D_MODEL];
    pooled[c] = s * (1.0f / (float)TT);
  }
  __syncthreads();
  if (tid < 128) {
    float s = bg1[tid];
    const float* w = Wg1 + (size_t)tid * D_MODEL;
    for (int i = 0; i < D_MODEL; ++i) s += pooled[i] * w[i];
    gh[tid] = s > 0.0f ? s : 0.0f;
  }
  __syncthreads();
  if (tid < NUM_EXPERTS) {
    float s = bg2[tid];
    const float* w = Wg2 + tid * 128;
    for (int i = 0; i < 128; ++i) s += gh[i] * w[i];
    gwl[tid] = s;
  }
  __syncthreads();
  if (tid == 0) {
    float mx = gwl[0];
    for (int e = 1; e < NUM_EXPERTS; ++e) mx = fmaxf(mx, gwl[e]);
    float sum = 0.0f, tmp[NUM_EXPERTS];
    for (int e = 0; e < NUM_EXPERTS; ++e) { tmp[e] = __expf(gwl[e] - mx); sum += tmp[e]; }
    const float inv = 1.0f / sum;
    for (int e = 0; e < NUM_EXPERTS; ++e) { gwl[e] = tmp[e] * inv; gw[b * NUM_EXPERTS + e] = gwl[e]; }
  }
  __syncthreads();
  const float* bptr[3] = { bq, bk, bv };
  for (int p = 0; p < 3; ++p) {
    const float* bb = bptr[p];
    for (int c = tid; c < D_MODEL; c += 256) {
      float s = 0.0f;
      for (int e = 0; e < NUM_EXPERTS; ++e) s += gwl[e] * bb[e * D_MODEL + c];
      beff[(size_t)(p * BB + b) * D_MODEL + c] = s;
    }
  }
}

// ---------------------------------------------------------------------------
// Kernel 2: fp32 -> bf16 (vectorized x4)
// ---------------------------------------------------------------------------
__global__ __launch_bounds__(256) void cvt_bf16_kernel(
    const float* __restrict__ in, unsigned short* __restrict__ out, size_t n4)
{
  size_t i = (size_t)blockIdx.x * 256 + threadIdx.x;
  if (i >= n4) return;
  float4 v = ((const float4*)in)[i];
  uint2 o;
  o.x = pack2bf(v.x, v.y);
  o.y = pack2bf(v.z, v.w);
  ((uint2*)out)[i] = o;
}

// ---------------------------------------------------------------------------
// Kernel 3: W_eff[p][b] = sum_e gw[b,e] * W_p[e]  (bf16 out)
// ---------------------------------------------------------------------------
__global__ __launch_bounds__(256) void weff_kernel(
    const float* __restrict__ Wq, const float* __restrict__ Wk,
    const float* __restrict__ Wv, const float* __restrict__ gw,
    unsigned short* __restrict__ weff)
{
  const size_t CC = (size_t)D_MODEL * D_MODEL;
  const size_t NCC4 = CC / 4;
  size_t idx = (size_t)blockIdx.x * 256 + threadIdx.x;   // 3*BB*NCC4 total
  const int p = (int)(idx / (BB * NCC4));
  size_t r = idx - (size_t)p * (BB * NCC4);
  const int b = (int)(r / NCC4);
  const size_t q4 = r - (size_t)b * NCC4;
  const float* W = (p == 0) ? Wq : (p == 1) ? Wk : Wv;
  float4 acc = make_float4(0.f, 0.f, 0.f, 0.f);
#pragma unroll
  for (int e = 0; e < NUM_EXPERTS; ++e) {
    const float g = gw[b * NUM_EXPERTS + e];
    const float4 w = *((const float4*)(W + (size_t)e * CC) + q4);
    acc.x += g * w.x; acc.y += g * w.y; acc.z += g * w.z; acc.w += g * w.w;
  }
  uint2 o;
  o.x = pack2bf(acc.x, acc.y);
  o.y = pack2bf(acc.z, acc.w);
  *((uint2*)(weff + (size_t)(p * BB + b) * CC + q4 * 4)) = o;
}

// ---------------------------------------------------------------------------
// Kernel 4: generic bf16 WMMA GEMM   D[z] = A[z%aMod] * W[z]^T + bias[z]
//   A: [M,K] bf16 row-major, W: [N,K] bf16 row-major, bias: [N] f32
//   block tile 128x128, K-step 32, 8 waves (each 32x64 -> 8 WMMA tiles)
//   If z/aMod == trPhase: store D transposed ([N][M], bf16, vectorized b128)
// ---------------------------------------------------------------------------
__global__ __launch_bounds__(256) void gemm_bf16_kernel(
    const unsigned short* __restrict__ Abase,
    const unsigned short* __restrict__ Wbase,
    const float* __restrict__ biasBase,
    void* __restrict__ Dbase,
    int M, int N, int K, int aMod, int outBf16, int trPhase)
{
#if HAVE_ASYNC_LDS
  __shared__ unsigned short Asm[2][128 * 40];   // 32 cols + 8 pad, rows 16B aligned
  __shared__ unsigned short Wsm[2][128 * 40];
#else
  __shared__ unsigned short Asm[1][128 * 40];
  __shared__ unsigned short Wsm[1][128 * 40];
#endif

  const int z = blockIdx.z;
  const unsigned short* A = Abase + (size_t)(z % aMod) * (size_t)M * K;
  const unsigned short* W = Wbase + (size_t)z * (size_t)N * K;
  const float* bias = biasBase + (size_t)z * N;
  const size_t dOff = (size_t)z * (size_t)M * N;
  const int doTr = ((z / aMod) == trPhase);

  const int tid = threadIdx.x, lane = tid & 31, wid = tid >> 5;
  const int waveM = wid & 3, waveN = wid >> 2;
  const int lm = lane & 15, hi = lane >> 4;
  const int kbA = hi ? 16 : 0;     // A frag K-half select
  const int kbB = hi ? 16 : 0;     // B frag K-half select
  const int ldRow = tid >> 1, ldHalf = tid & 1;
  const size_t aRow = (size_t)blockIdx.y * 128 + ldRow;
  const size_t wRow = (size_t)blockIdx.x * 128 + ldRow;
  const unsigned short* gA = A + aRow * K + ldHalf * 16;
  const unsigned short* gW = W + wRow * K + ldHalf * 16;
  const int ldsOff = ldRow * 40 + ldHalf * 16;
  const int NK = K / 32;

  v8f acc[2][4] = {};

#if HAVE_ASYNC_LDS
  {
    ASYNC_CP16(gA, &Asm[0][ldsOff], 0);
    ASYNC_CP16(gA, &Asm[0][ldsOff], 16);
    ASYNC_CP16(gW, &Wsm[0][ldsOff], 0);
    ASYNC_CP16(gW, &Wsm[0][ldsOff], 16);
    wait_async0();
    __syncthreads();
  }
#endif

  for (int kt = 0; kt < NK; ++kt) {
    int cur;
#if HAVE_ASYNC_LDS
    cur = kt & 1;
    if (kt + 1 < NK) {
      const unsigned short* ga2 = gA + (size_t)(kt + 1) * 32;
      const unsigned short* gw2 = gW + (size_t)(kt + 1) * 32;
      unsigned short* la = &Asm[cur ^ 1][ldsOff];
      unsigned short* lw = &Wsm[cur ^ 1][ldsOff];
      ASYNC_CP16(ga2, la, 0);
      ASYNC_CP16(ga2, la, 16);
      ASYNC_CP16(gw2, lw, 0);
      ASYNC_CP16(gw2, lw, 16);
    }
#else
    cur = 0;
    const uint4* ga4 = (const uint4*)(gA + (size_t)kt * 32);
    uint4 a0 = ga4[0], a1 = ga4[1];
    const uint4* gw4 = (const uint4*)(gW + (size_t)kt * 32);
    uint4 w0 = gw4[0], w1 = gw4[1];
    __syncthreads();
    uint4* as = (uint4*)(&Asm[0][ldsOff]);
    as[0] = a0; as[1] = a1;
    uint4* ws = (uint4*)(&Wsm[0][ldsOff]);
    ws[0] = w0; ws[1] = w1;
    __syncthreads();
#endif

    FragB16 af[2], bfr[4];
#pragma unroll
    for (int mi = 0; mi < 2; ++mi) {
      const unsigned short* ap = &Asm[cur][(waveM * 32 + mi * 16 + lm) * 40 + kbA];
#pragma unroll
      for (int j = 0; j < 4; ++j) {
        af[mi].u[j]     = *(const unsigned*)(ap + 2 * j);
        af[mi].u[4 + j] = *(const unsigned*)(ap + 8 + 2 * j);
      }
    }
#pragma unroll
    for (int ni = 0; ni < 4; ++ni) {
      const unsigned short* wp = &Wsm[cur][(waveN * 64 + ni * 16 + lm) * 40 + kbB];
#pragma unroll
      for (int j = 0; j < 8; ++j) bfr[ni].u[j] = *(const unsigned*)(wp + 2 * j);
    }
#pragma unroll
    for (int mi = 0; mi < 2; ++mi)
#pragma unroll
      for (int ni = 0; ni < 4; ++ni)
        acc[mi][ni] = __builtin_amdgcn_wmma_f32_16x16x32_bf16(
            false, af[mi].v, false, bfr[ni].v, (short)0, acc[mi][ni], false, false);

#if HAVE_ASYNC_LDS
    wait_async0();
    __syncthreads();
#endif
  }

#pragma unroll
  for (int mi = 0; mi < 2; ++mi) {
#pragma unroll
    for (int ni = 0; ni < 4; ++ni) {
      const int gcol = blockIdx.x * 128 + waveN * 64 + ni * 16 + lm;
      const float bval = bias[gcol];
      if (doTr) {
        // transposed bf16 store: 8 consecutive M elements -> one b128
        const size_t growBase = (size_t)blockIdx.y * 128 + waveM * 32 + mi * 16 + 8 * hi;
        uint4 pk;
        pk.x = pack2bf(acc[mi][ni][0] + bval, acc[mi][ni][1] + bval);
        pk.y = pack2bf(acc[mi][ni][2] + bval, acc[mi][ni][3] + bval);
        pk.z = pack2bf(acc[mi][ni][4] + bval, acc[mi][ni][5] + bval);
        pk.w = pack2bf(acc[mi][ni][6] + bval, acc[mi][ni][7] + bval);
        *(uint4*)((unsigned short*)Dbase + dOff + (size_t)gcol * M + growBase) = pk;
      } else {
#pragma unroll
        for (int r = 0; r < 8; ++r) {
          const size_t grow = (size_t)blockIdx.y * 128 + waveM * 32 + mi * 16 + r + 8 * hi;
          const float v = acc[mi][ni][r] + bval;
          if (outBf16) ((unsigned short*)Dbase)[dOff + grow * (size_t)N + gcol] = f2bf(v);
          else         ((float*)Dbase)[dOff + grow * (size_t)N + gcol] = v;
        }
      }
    }
  }
}

// ---------------------------------------------------------------------------
// Kernel 5: flash attention, one (b,h) per workgroup, 16 q-rows per wave
//   Q,K: [B,T,C] bf16;  Vt: [B,C,T] bf16 (transposed by QKV GEMM)
// ---------------------------------------------------------------------------
__global__ __launch_bounds__(256) void attn_kernel(
    const unsigned short* __restrict__ Q,
    const unsigned short* __restrict__ Kx,
    const unsigned short* __restrict__ Vt,
    unsigned short* __restrict__ Out)
{
#if HAVE_ASYNC_LDS
  __shared__ unsigned short Ksm[2][32 * 72];   // [key][d], rows 144B (16B aligned)
  __shared__ unsigned short Vsm[2][64 * 40];   // [dh][key], rows 80B (16B aligned)
#else
  __shared__ unsigned short Ksm[1][32 * 72];
  __shared__ unsigned short Vsm[1][64 * 40];
#endif
  __shared__ unsigned short Psm[8][16 * 36];   // per-wave P tile 16x32

  const int tid = threadIdx.x, lane = tid & 31, wid = tid >> 5;
  const int bh = blockIdx.y;
  const int b = bh >> 4, h = bh & 15;
  const int q0 = blockIdx.x * 128 + wid * 16;

  const int lm = lane & 15;
  const int hi = lane >> 4;
  const int kbA = hi ? 16 : 0;
  const int kbB = hi ? 16 : 0;

  // Q fragments (A 16x32) for d=[0,32) and [32,64), loaded straight from global
  FragB16 qf[2];
  {
    const unsigned short* qp = Q + ((size_t)b * TT + q0 + lm) * D_MODEL + h * HEAD_DIM;
#pragma unroll
    for (int s = 0; s < 2; ++s) {
      const unsigned short* p0 = qp + s * 32 + kbA;
#pragma unroll
      for (int j = 0; j < 4; ++j) {
        qf[s].u[j]     = *(const unsigned*)(p0 + 2 * j);
        qf[s].u[4 + j] = *(const unsigned*)(p0 + 8 + 2 * j);
      }
    }
  }

  v8f oacc[4] = {};
  float mrun[8], lrun[8];
#pragma unroll
  for (int r = 0; r < 8; ++r) { mrun[r] = -3.0e38f; lrun[r] = 0.0f; }

  // staging addresses: one 16B chunk per thread per matrix
  const int kRow = tid >> 3, kSeg = tid & 7;          // 32 keys x 8 segs
  const int vRow = tid >> 2, vSeg = tid & 3;          // 64 dh  x 4 segs
  const unsigned short* gK =
      Kx + ((size_t)b * TT + kRow) * D_MODEL + h * HEAD_DIM + kSeg * 8;
  const unsigned short* gV =
      Vt + (size_t)b * D_MODEL * TT + (size_t)(h * HEAD_DIM + vRow) * TT + vSeg * 8;
  const int kOff = kRow * 72 + kSeg * 8;
  const int vOff = vRow * 40 + vSeg * 8;
  const int NKB = TT / 32;

#if HAVE_ASYNC_LDS
  {
    ASYNC_CP16(gK, &Ksm[0][kOff], 0);
    ASYNC_CP16(gV, &Vsm[0][vOff], 0);
    wait_async0();
    __syncthreads();
  }
#endif

  for (int kb32 = 0; kb32 < NKB; ++kb32) {
    int cur;
#if HAVE_ASYNC_LDS
    cur = kb32 & 1;
    if (kb32 + 1 < NKB) {
      ASYNC_CP16(gK + (size_t)(kb32 + 1) * 32 * D_MODEL, &Ksm[cur ^ 1][kOff], 0);
      ASYNC_CP16(gV + (size_t)(kb32 + 1) * 32,           &Vsm[cur ^ 1][vOff], 0);
    }
#else
    cur = 0;
    uint4 kv = *(const uint4*)(gK + (size_t)kb32 * 32 * D_MODEL);
    uint4 vv = *(const uint4*)(gV + (size_t)kb32 * 32);
    __syncthreads();
    *(uint4*)(&Ksm[0][kOff]) = kv;
    *(uint4*)(&Vsm[0][vOff]) = vv;
    __syncthreads();
#endif

    // S = Q K^T for two 16-key subtiles (K-dim = 64 -> 2 wmma each)
    v8f sacc[2] = {};
#pragma unroll
    for (int s = 0; s < 2; ++s) {
#pragma unroll
      for (int dstep = 0; dstep < 2; ++dstep) {
        FragB16 kf;
        const unsigned short* kp = &Ksm[cur][(s * 16 + lm) * 72 + dstep * 32 + kbB];
#pragma unroll
        for (int j = 0; j < 8; ++j) kf.u[j] = *(const unsigned*)(kp + 2 * j);
        sacc[s] = __builtin_amdgcn_wmma_f32_16x16x32_bf16(
            false, qf[dstep].v, false, kf.v, (short)0, sacc[s], false, false);
      }
    }

    // online softmax per row (rows m = r + 8*hi); write P tile to LDS as bf16
#pragma unroll
    for (int r = 0; r < 8; ++r) {
      float s0 = sacc[0][r] * 0.125f;   // 1/sqrt(64)
      float s1 = sacc[1][r] * 0.125f;
      float mx = fmaxf(s0, s1);
#pragma unroll
      for (int d = 1; d < 16; d <<= 1) mx = fmaxf(mx, __shfl_xor(mx, d, 32));
      const float mnew = fmaxf(mrun[r], mx);
      const float alpha = __expf(mrun[r] - mnew);
      const float p0 = __expf(s0 - mnew);
      const float p1 = __expf(s1 - mnew);
      float rs = p0 + p1;
#pragma unroll
      for (int d = 1; d < 16; d <<= 1) rs += __shfl_xor(rs, d, 32);
      lrun[r] = lrun[r] * alpha + rs;
      mrun[r] = mnew;
#pragma unroll
      for (int n = 0; n < 4; ++n) oacc[n][r] *= alpha;
      const int mrow = r + 8 * hi;
      Psm[wid][mrow * 36 + lm]      = f2bf(p0);
      Psm[wid][mrow * 36 + 16 + lm] = f2bf(p1);
    }
    asm volatile("s_wait_dscnt 0x0" ::: "memory");   // wave-local P write->read ordering

    // P fragment (A 16x32)
    FragB16 pf;
    {
      const unsigned short* pp = &Psm[wid][lm * 36 + kbA];
#pragma unroll
      for (int j = 0; j < 4; ++j) {
        pf.u[j]     = *(const unsigned*)(pp + 2 * j);
        pf.u[4 + j] = *(const unsigned*)(pp + 8 + 2 * j);
      }
    }

    // O += P * V : B-frag contiguous reads from transposed V tile
#pragma unroll
    for (int n = 0; n < 4; ++n) {
      FragB16 vf;
      const unsigned short* vp = &Vsm[cur][(n * 16 + lm) * 40 + kbB];
#pragma unroll
      for (int j = 0; j < 8; ++j) vf.u[j] = *(const unsigned*)(vp + 2 * j);
      oacc[n] = __builtin_amdgcn_wmma_f32_16x16x32_bf16(
          false, pf.v, false, vf.v, (short)0, oacc[n], false, false);
    }

#if HAVE_ASYNC_LDS
    wait_async0();
    __syncthreads();
#endif
  }

  // normalize and write bf16 [B,T,C]
#pragma unroll
  for (int r = 0; r < 8; ++r) {
    const float inv = 1.0f / lrun[r];
    const size_t row = (size_t)b * TT + q0 + r + 8 * hi;
#pragma unroll
    for (int n = 0; n < 4; ++n)
      Out[row * D_MODEL + h * HEAD_DIM + n * 16 + lm] = f2bf(oacc[n][r] * inv);
  }
}

// ---------------------------------------------------------------------------
extern "C" void kernel_launch(void* const* d_in, const int* in_sizes, int n_in,
                              void* d_out, int out_size, void* d_ws, size_t ws_size,
                              hipStream_t stream) {
  const float* x   = (const float*)d_in[0];
  const float* Wq  = (const float*)d_in[1];
  const float* bq  = (const float*)d_in[2];
  const float* Wk  = (const float*)d_in[3];
  const float* bk  = (const float*)d_in[4];
  const float* Wv  = (const float*)d_in[5];
  const float* bv  = (const float*)d_in[6];
  const float* Wg1 = (const float*)d_in[7];
  const float* bg1 = (const float*)d_in[8];
  const float* Wg2 = (const float*)d_in[9];
  const float* bg2 = (const float*)d_in[10];
  const float* Wo  = (const float*)d_in[11];
  const float* bo  = (const float*)d_in[12];

  char* ws = (char*)d_ws;
  size_t off = 0;
  auto take = [&](size_t bytes) { char* p = ws + off; off = (off + bytes + 255) & ~(size_t)255; return p; };
  float* gw            = (float*)take((size_t)BB * NUM_EXPERTS * sizeof(float));
  float* beff          = (float*)take((size_t)3 * BB * D_MODEL * sizeof(float));
  unsigned short* xb   = (unsigned short*)take((size_t)BB * TT * D_MODEL * 2);
  unsigned short* wob  = (unsigned short*)take((size_t)D_MODEL * D_MODEL * 2);
  unsigned short* weff = (unsigned short*)take((size_t)3 * BB * D_MODEL * D_MODEL * 2);
  unsigned short* qkv  = (unsigned short*)take((size_t)3 * BB * TT * D_MODEL * 2);
  unsigned short* attn = (unsigned short*)take((size_t)BB * TT * D_MODEL * 2);

  // 1. gating + effective biases
  gate_kernel<<<BB, 256, 0, stream>>>(x, Wg1, bg1, Wg2, bg2, bq, bk, bv, gw, beff);

  // 2. bf16 conversions
  {
    size_t n4 = (size_t)BB * TT * D_MODEL / 4;
    cvt_bf16_kernel<<<(unsigned)((n4 + 255) / 256), 256, 0, stream>>>(x, xb, n4);
    size_t w4 = (size_t)D_MODEL * D_MODEL / 4;
    cvt_bf16_kernel<<<(unsigned)((w4 + 255) / 256), 256, 0, stream>>>(Wo, wob, w4);
  }

  // 3. mixed expert weights (bf16)
  weff_kernel<<<(unsigned)((size_t)3 * BB * (D_MODEL * (size_t)D_MODEL / 4) / 256),
               256, 0, stream>>>(Wq, Wk, Wv, gw, weff);

  // 4. QKV projections: z = p*BB + b; V phase (z/BB == 2) stored transposed
  gemm_bf16_kernel<<<dim3(D_MODEL / 128, TT / 128, 3 * BB), 256, 0, stream>>>(
      xb, weff, beff, qkv, TT, D_MODEL, D_MODEL, /*aMod=*/BB, /*outBf16=*/1, /*trPhase=*/2);

  // 5. attention (V read as [B,C,T])
  attn_kernel<<<dim3(TT / 128, BB * N_HEADS), 256, 0, stream>>>(
      qkv,
      qkv + (size_t)BB * TT * D_MODEL,
      qkv + (size_t)2 * BB * TT * D_MODEL,
      attn);

  // 6. output projection (fp32 result)
  gemm_bf16_kernel<<<dim3(D_MODEL / 128, (BB * TT) / 128, 1), 256, 0, stream>>>(
      attn, wob, bo, d_out, BB * TT, D_MODEL, D_MODEL, /*aMod=*/1, /*outBf16=*/0, /*trPhase=*/-1);
}
